// MultiHeadSelfAttention_47047071760626
// MI455X (gfx1250) — matrix operand order repaired and checked by reading, not compiled, not run on hardware
//
#include <hip/hip_runtime.h>

// ---------------------------------------------------------------------------
// MI455X (gfx1250) multi-head self-attention forward.
// bf16 WMMA (V_WMMA_F32_16X16X32_BF16), f32 accumulation, wave32.
// ---------------------------------------------------------------------------

typedef __attribute__((ext_vector_type(16))) __bf16 v16bf;
typedef __attribute__((ext_vector_type(8)))  __bf16 v8bf;
typedef __attribute__((ext_vector_type(8)))  float  v8f;
typedef int v4i_g __attribute__((vector_size(16)));   // matches builtin param type

#define B_DIM 2
#define S_DIM 2048
#define D_MODEL 1024
#define H_DIM 16
#define DH 64
#define LN_THETA 9.2103403719761836f   // ln(10000)
#define NEG_INF (-3.0e38f)

#define AS1 __attribute__((address_space(1)))
#define AS3 __attribute__((address_space(3)))

#if __has_builtin(__builtin_amdgcn_global_load_async_to_lds_b128)
#define HAVE_ASYNC_LDS 1
#else
#define HAVE_ASYNC_LDS 0
#endif

// Build a 16-element bf16 fragment from two 16-byte loads (global or LDS).
__device__ __forceinline__ v16bf ld_frag16(const __bf16* p0, const __bf16* p1) {
    v8bf lo = *(const v8bf*)p0;
    v8bf hi = *(const v8bf*)p1;
    return __builtin_shufflevector(lo, hi, 0,1,2,3,4,5,6,7,8,9,10,11,12,13,14,15);
}

__device__ __forceinline__ v8f wmma_bf16(v16bf a, v16bf b, v8f c) {
    // 8 args: (neg_a, A, neg_b, B, c_mod, C, reuse_a, reuse_b)
    return __builtin_amdgcn_wmma_f32_16x16x32_bf16(false, a, false, b, (short)0, c,
                                                   false, false);
}

// 16-byte global -> LDS copy: async path (GLOBAL_LOAD_ASYNC_TO_LDS_B128).
__device__ __forceinline__ void copy16_g2l(__bf16* ldst, const __bf16* gsrc) {
#if HAVE_ASYNC_LDS
    __builtin_amdgcn_global_load_async_to_lds_b128(
        (AS1 v4i_g*)(__bf16*)gsrc, (AS3 v4i_g*)ldst, 0, 0);
#else
    *(uint4*)ldst = *(const uint4*)gsrc;
#endif
}

__device__ __forceinline__ void wait_async_lds() {
#if HAVE_ASYNC_LDS
#if __has_builtin(__builtin_amdgcn_s_wait_asynccnt)
    __builtin_amdgcn_s_wait_asynccnt(0);
#else
    asm volatile("s_wait_asynccnt 0x0" ::: "memory");
#endif
#endif
}

// ---------------------------------------------------------------------------
// Kernel 0: f32 -> bf16 conversion (x and the four weight matrices)
// ---------------------------------------------------------------------------
__global__ void cvt_f32_bf16_kernel(const float* __restrict__ src,
                                    __bf16* __restrict__ dst, int n) {
    int i = blockIdx.x * blockDim.x + threadIdx.x;
    if (i < n) dst[i] = (__bf16)src[i];
}

// ---------------------------------------------------------------------------
// Kernel 1: fused QKV projection + RoPE.  Each wave: 32(M) x 64(N) tile,
// 8 WMMAs per 32-deep K-step; A fragments reused 4x, B fragments 2x.
//   blockIdx.x : m-tile (32 rows of B*S)        [128]
//   blockIdx.y : n-group (8 waves * 64 cols)    [2]
//   blockIdx.z : 0=Q, 1=K, 2=V                  [3]
// ---------------------------------------------------------------------------
__global__ void qkv_proj_kernel(const __bf16* __restrict__ xh,
                                const __bf16* __restrict__ Wqh,
                                const __bf16* __restrict__ Wkh,
                                const __bf16* __restrict__ Wvh,
                                const int* __restrict__ tpos,
                                __bf16* __restrict__ Qh,   // [B][H][S][DH]
                                __bf16* __restrict__ Kh,   // [B][H][S][DH]
                                __bf16* __restrict__ Vt) { // [B][H][DH][S]
    const int lane = threadIdx.x & 31;
    const int wave = threadIdx.x >> 5;
    const int l15  = lane & 15;
    const int h2   = lane >> 4;
    const int m0   = blockIdx.x * 32;
    const int n0   = (blockIdx.y * 8 + wave) * 64;
    const int z    = blockIdx.z;

    const __bf16* Wh = (z == 0) ? Wqh : (z == 1) ? Wkh : Wvh;

    const __bf16* arow0 = xh + (size_t)(m0 + l15) * D_MODEL;
    const __bf16* arow1 = arow0 + (size_t)16 * D_MODEL;
    const __bf16* brow  = Wh + (size_t)(n0 + l15) * D_MODEL;

    v8f acc[2][4] = {};
    for (int k0 = 0; k0 < D_MODEL; k0 += 32) {
        v16bf a0 = ld_frag16(arow0 + k0 + 8 * h2, arow0 + k0 + 16 + 8 * h2);
        v16bf a1 = ld_frag16(arow1 + k0 + 8 * h2, arow1 + k0 + 16 + 8 * h2);
        #pragma unroll
        for (int ns = 0; ns < 4; ++ns) {
            const __bf16* br = brow + (size_t)ns * 16 * D_MODEL;
            v16bf bb = ld_frag16(br + k0 + 8 * h2, br + k0 + 16 + 8 * h2);
            acc[0][ns] = wmma_bf16(a0, bb, acc[0][ns]);
            acc[1][ns] = wmma_bf16(a1, bb, acc[1][ns]);
        }
    }

    #pragma unroll
    for (int ns = 0; ns < 4; ++ns) {
        const int n = n0 + ns * 16 + l15;   // absolute output column
        const int h = n >> 6;               // head
        const int d = n & 63;               // dim within head
        if (z < 2) {
            // RoPE: even/odd partner lives in the adjacent lane (column n^1).
            const float inv_freq = expf(-(float)(d & ~1) * (LN_THETA / (float)DH));
            const float sgn = (d & 1) ? 1.0f : -1.0f;
            __bf16* Out = (z == 0) ? Qh : Kh;
            #pragma unroll
            for (int ms = 0; ms < 2; ++ms) {
                #pragma unroll
                for (int r = 0; r < 8; ++r) {
                    int m  = m0 + ms * 16 + r + 8 * h2;   // row in B*S
                    int bb = m >> 11;                     // S = 2048
                    int s  = m & (S_DIM - 1);
                    float ang = (float)tpos[s] * inv_freq;
                    float c = cosf(ang), sn = sinf(ang);
                    float v = acc[ms][ns][r];
                    float p = __shfl_xor(v, 1, 32);       // partner column value
                    float o = v * c + sgn * p * sn;
                    Out[(((size_t)bb * H_DIM + h) * S_DIM + s) * DH + d] = (__bf16)o;
                }
            }
        } else {
            // V stored transposed: Vt[b][h][d][s]
            #pragma unroll
            for (int ms = 0; ms < 2; ++ms) {
                #pragma unroll
                for (int r = 0; r < 8; ++r) {
                    int m  = m0 + ms * 16 + r + 8 * h2;
                    int bb = m >> 11;
                    int s  = m & (S_DIM - 1);
                    Vt[(((size_t)bb * H_DIM + h) * DH + d) * S_DIM + s] =
                        (__bf16)acc[ms][ns][r];
                }
            }
        }
    }
}

// ---------------------------------------------------------------------------
// Kernel 2: causal flash attention.
// 128-thread block = 4 waves = 4 consecutive 16-row q-tiles of one (b,h).
// K-tile (32x64) and V-tile (64x32) are staged in LDS once per 32-key step
// (async global->LDS when available) and shared by all 4 waves.
// ---------------------------------------------------------------------------
__global__ void flash_attn_kernel(const __bf16* __restrict__ Qh,
                                  const __bf16* __restrict__ Kh,
                                  const __bf16* __restrict__ Vt,
                                  __bf16* __restrict__ Oh) { // [B][S][D_MODEL]
    __shared__ __align__(16) __bf16 Ktile[32 * 64];      // [j_local][d]
    __shared__ __align__(16) __bf16 Vtile[64 * 32];      // [d][j_local]
    __shared__ __align__(16) __bf16 plds[4][16 * 40];    // per-wave P tile

    const int tid  = threadIdx.x;
    const int lane = tid & 31;
    const int wave = tid >> 5;
    const int l15  = lane & 15;
    const int h2   = lane >> 4;
    const int q0b  = blockIdx.x * 64;
    const int q0   = q0b + wave * 16;
    const int h    = blockIdx.y;
    const int b    = blockIdx.z;

    const __bf16* Qb = Qh + ((size_t)b * H_DIM + h) * (size_t)S_DIM * DH;
    const __bf16* Kb = Kh + ((size_t)b * H_DIM + h) * (size_t)S_DIM * DH;
    const __bf16* Vb = Vt + ((size_t)b * H_DIM + h) * (size_t)DH * S_DIM;

    // Q fragments: rows q0..q0+15, d split into two K-steps of 32.
    const __bf16* qrow = Qb + (size_t)(q0 + l15) * DH;
    v16bf qf0 = ld_frag16(qrow +  0 + 16 * h2, qrow +  0 + 16 * h2 + 8);
    v16bf qf1 = ld_frag16(qrow + 32 + 16 * h2, qrow + 32 + 16 * h2 + 8);

    v8f oacc[4] = {};
    float mrow[8], lrow[8];
    #pragma unroll
    for (int r = 0; r < 8; ++r) { mrow[r] = NEG_INF; lrow[r] = 0.0f; }

    const int jend = q0b + 64;            // uniform causal bound for the block
    for (int j0 = 0; j0 < jend; j0 += 32) {
        __syncthreads();                  // protect tiles from previous readers

        // ---- cooperative K/V tile fill: 2x 16B chunks per thread per tile ----
        #pragma unroll
        for (int c = tid; c < 256; c += 128) {     // K: 32 rows x 128B
            const int row = c >> 3, col = (c & 7) * 8;
            copy16_g2l(&Ktile[row * 64 + col], Kb + (size_t)(j0 + row) * DH + col);
        }
        #pragma unroll
        for (int c = tid; c < 256; c += 128) {     // V: 64 rows x 64B
            const int row = c >> 2, col = (c & 3) * 8;
            copy16_g2l(&Vtile[row * 32 + col], Vb + (size_t)row * S_DIM + j0 + col);
        }
        wait_async_lds();
        __syncthreads();

        // ---- S = Q K^T for 32 keys (two 16-wide column sub-tiles) ----
        const __bf16* kr0 = &Ktile[(size_t)l15 * 64];
        const __bf16* kr1 = &Ktile[(size_t)(16 + l15) * 64];
        v16bf b00 = ld_frag16(kr0 +  0 + 16 * h2, kr0 +  0 + 16 * h2 + 8);
        v16bf b01 = ld_frag16(kr0 + 32 + 16 * h2, kr0 + 32 + 16 * h2 + 8);
        v16bf b10 = ld_frag16(kr1 +  0 + 16 * h2, kr1 +  0 + 16 * h2 + 8);
        v16bf b11 = ld_frag16(kr1 + 32 + 16 * h2, kr1 + 32 + 16 * h2 + 8);

        v8f s0 = {}, s1 = {};
        s0 = wmma_bf16(qf0, b00, s0);
        s0 = wmma_bf16(qf1, b01, s0);
        s1 = wmma_bf16(qf0, b10, s1);
        s1 = wmma_bf16(qf1, b11, s1);

        // ---- causal mask + online softmax (row reductions over 16 lanes) ----
        const int jc0 = j0 + l15;
        const int jc1 = j0 + 16 + l15;
        #pragma unroll
        for (int r = 0; r < 8; ++r) {
            const int qr = q0 + r + 8 * h2;
            float v0 = (jc0 <= qr) ? s0[r] * 0.125f : NEG_INF;  // 1/sqrt(64)
            float v1 = (jc1 <= qr) ? s1[r] * 0.125f : NEG_INF;

            float mx = fmaxf(v0, v1);
            mx = fmaxf(mx, __shfl_xor(mx, 1, 32));
            mx = fmaxf(mx, __shfl_xor(mx, 2, 32));
            mx = fmaxf(mx, __shfl_xor(mx, 4, 32));
            mx = fmaxf(mx, __shfl_xor(mx, 8, 32));
            float mn = fmaxf(mrow[r], mx);

            float scale = expf(mrow[r] - mn);
            float p0 = expf(v0 - mn);
            float p1 = expf(v1 - mn);
            float ps = p0 + p1;
            ps += __shfl_xor(ps, 1, 32);
            ps += __shfl_xor(ps, 2, 32);
            ps += __shfl_xor(ps, 4, 32);
            ps += __shfl_xor(ps, 8, 32);

            lrow[r] = lrow[r] * scale + ps;
            mrow[r] = mn;
            oacc[0][r] *= scale; oacc[1][r] *= scale;
            oacc[2][r] *= scale; oacc[3][r] *= scale;

            const int m = r + 8 * h2;     // row within this wave's P tile
            plds[wave][m * 40 + l15]      = (__bf16)p0;
            plds[wave][m * 40 + 16 + l15] = (__bf16)p1;
        }

        // all 32 lanes' P writes must land before cross-lane A-layout reads
        asm volatile("s_wait_dscnt 0x0" ::: "memory");

        // ---- P fragment (A layout, 16x32 over j) from this wave's LDS tile ----
        const __bf16* prow = &plds[wave][(size_t)l15 * 40];
        v16bf pf = ld_frag16(prow + 16 * h2, prow + 16 * h2 + 8);

        // ---- O += P * V (V tile rows contiguous over j) ----
        #pragma unroll
        for (int dsub = 0; dsub < 4; ++dsub) {
            const __bf16* vrow = &Vtile[(size_t)(dsub * 16 + l15) * 32 + 16 * h2];
            v16bf vf = ld_frag16(vrow, vrow + 8);
            oacc[dsub] = wmma_bf16(pf, vf, oacc[dsub]);
        }
    }

    // ---- normalize and store O (layout [b][s][h*64+d]) ----
    #pragma unroll
    for (int r = 0; r < 8; ++r) {
        const float inv = 1.0f / lrow[r];
        const size_t row = (size_t)b * S_DIM + (q0 + r + 8 * h2);
        __bf16* outp = Oh + row * D_MODEL + h * DH + l15;
        outp[0]  = (__bf16)(oacc[0][r] * inv);
        outp[16] = (__bf16)(oacc[1][r] * inv);
        outp[32] = (__bf16)(oacc[2][r] * inv);
        outp[48] = (__bf16)(oacc[3][r] * inv);
    }
}

// ---------------------------------------------------------------------------
// Kernel 3: output projection  out = O @ Wo^T  (f32 result), 32x64 wave tiles.
// ---------------------------------------------------------------------------
__global__ void out_proj_kernel(const __bf16* __restrict__ Oh,
                                const __bf16* __restrict__ Woh,
                                float* __restrict__ out) {
    const int lane = threadIdx.x & 31;
    const int wave = threadIdx.x >> 5;
    const int l15  = lane & 15;
    const int h2   = lane >> 4;
    const int m0   = blockIdx.x * 32;
    const int n0   = (blockIdx.y * 8 + wave) * 64;

    const __bf16* arow0 = Oh + (size_t)(m0 + l15) * D_MODEL;
    const __bf16* arow1 = arow0 + (size_t)16 * D_MODEL;
    const __bf16* brow  = Woh + (size_t)(n0 + l15) * D_MODEL;

    v8f acc[2][4] = {};
    for (int k0 = 0; k0 < D_MODEL; k0 += 32) {
        v16bf a0 = ld_frag16(arow0 + k0 + 8 * h2, arow0 + k0 + 16 + 8 * h2);
        v16bf a1 = ld_frag16(arow1 + k0 + 8 * h2, arow1 + k0 + 16 + 8 * h2);
        #pragma unroll
        for (int ns = 0; ns < 4; ++ns) {
            const __bf16* br = brow + (size_t)ns * 16 * D_MODEL;
            v16bf bb = ld_frag16(br + k0 + 8 * h2, br + k0 + 16 + 8 * h2);
            acc[0][ns] = wmma_bf16(a0, bb, acc[0][ns]);
            acc[1][ns] = wmma_bf16(a1, bb, acc[1][ns]);
        }
    }

    #pragma unroll
    for (int ms = 0; ms < 2; ++ms) {
        #pragma unroll
        for (int ns = 0; ns < 4; ++ns) {
            #pragma unroll
            for (int r = 0; r < 8; ++r) {
                const size_t m = (size_t)(m0 + ms * 16 + r + 8 * h2);
                out[m * D_MODEL + n0 + ns * 16 + l15] = acc[ms][ns][r];
            }
        }
    }
}

// ---------------------------------------------------------------------------
// Host-side launch
// ---------------------------------------------------------------------------
extern "C" void kernel_launch(void* const* d_in, const int* in_sizes, int n_in,
                              void* d_out, int out_size, void* d_ws, size_t ws_size,
                              hipStream_t stream) {
    const float* x  = (const float*)d_in[0];
    const float* Wq = (const float*)d_in[1];
    const float* Wk = (const float*)d_in[2];
    const float* Wv = (const float*)d_in[3];
    const float* Wo = (const float*)d_in[4];
    const int* tpos = (const int*)d_in[5];

    char* ws = (char*)d_ws;
    const size_t MiB = 1u << 20;
    __bf16* xh  = (__bf16*)(ws + 0 * MiB);    // 8 MiB  : x in bf16
    __bf16* Wqh = (__bf16*)(ws + 8 * MiB);    // 2 MiB
    __bf16* Wkh = (__bf16*)(ws + 10 * MiB);   // 2 MiB
    __bf16* Wvh = (__bf16*)(ws + 12 * MiB);   // 2 MiB
    __bf16* Woh = (__bf16*)(ws + 14 * MiB);   // 2 MiB
    __bf16* Qh  = (__bf16*)(ws + 16 * MiB);   // 8 MiB  : [B][H][S][DH]
    __bf16* Kh  = (__bf16*)(ws + 24 * MiB);   // 8 MiB  : [B][H][S][DH]
    __bf16* Vt  = (__bf16*)(ws + 32 * MiB);   // 8 MiB  : [B][H][DH][S]
    __bf16* Oh  = (__bf16*)(ws + 40 * MiB);   // 8 MiB  : [B][S][D]

    const int NX = B_DIM * S_DIM * D_MODEL;   // 4,194,304
    const int NW = D_MODEL * D_MODEL;         // 1,048,576

    cvt_f32_bf16_kernel<<<(NX + 255) / 256, 256, 0, stream>>>(x,  xh,  NX);
    cvt_f32_bf16_kernel<<<(NW + 255) / 256, 256, 0, stream>>>(Wq, Wqh, NW);
    cvt_f32_bf16_kernel<<<(NW + 255) / 256, 256, 0, stream>>>(Wk, Wkh, NW);
    cvt_f32_bf16_kernel<<<(NW + 255) / 256, 256, 0, stream>>>(Wv, Wvh, NW);
    cvt_f32_bf16_kernel<<<(NW + 255) / 256, 256, 0, stream>>>(Wo, Woh, NW);

    // QKV projection + RoPE: 128 m-tiles x (2 groups * 8 waves * 64 cols) x 3
    qkv_proj_kernel<<<dim3(128, 2, 3), 256, 0, stream>>>(xh, Wqh, Wkh, Wvh, tpos,
                                                         Qh, Kh, Vt);

    // Flash attention: 4 waves / block, 4 q-tiles sharing K/V via LDS
    flash_attn_kernel<<<dim3(S_DIM / 64, H_DIM, B_DIM), 128, 0, stream>>>(Qh, Kh, Vt, Oh);

    // Output projection -> f32
    out_proj_kernel<<<dim3(128, 2), 256, 0, stream>>>(Oh, Woh, (float*)d_out);
}
